// MegaCRN_62758062129204
// MI455X (gfx1250) — matrix operand order repaired and checked by reading, not compiled, not run on hardware
//
#include <hip/hip_runtime.h>
#include <math.h>

// ---------------- problem constants (from reference) ----------------
#define Bsz   8
#define Tsz   24
#define Nsz   1000
#define Cin   2
#define Hsz   64
#define Msz   64
#define HEADS 8
#define DH    8
#define BT    (Bsz*Tsz)          // 192
#define NPAD  1024               // padded node count (rows & K) for WMMA
#define MT    63                 // ceil(1000/16) output row tiles
#define TILES (BT*MT)            // 12096 wave-tiles (k_gout tiling)
#define WPB   8                  // waves per block (256 threads)
#define CSTRIDE 80               // LDS bytes per staged column (16B aligned, low-conflict)

typedef __attribute__((ext_vector_type(16))) __bf16 v16bf;
typedef __attribute__((ext_vector_type(8)))  float  v8f;

// ---------------- bf16 helpers (ushort storage) ----------------
__device__ __forceinline__ unsigned short f2bf(float f) {
    unsigned int u = __float_as_uint(f);
    u = u + 0x7FFFu + ((u >> 16) & 1u);          // round-to-nearest-even
    return (unsigned short)(u >> 16);
}
__device__ __forceinline__ float bf2f(unsigned short s) {
    return __uint_as_float(((unsigned int)s) << 16);
}

union AFrag { unsigned int u[8]; v16bf v; };

// A-fragment K offset for 16x32 bf16 A tile: VGPR j, lane-half kh
__device__ __forceinline__ int a_k0(int j, int kh) {
    return (j < 4) ? (kh * 8 + 2 * j) : (16 + kh * 8 + 2 * (j - 4));
}

// ============ kernel: convert supports to padded bf16 ============
__global__ void k_convert_s(const float* __restrict__ s1, const float* __restrict__ s2,
                            unsigned short* __restrict__ dst) {
    int idx = blockIdx.x * blockDim.x + threadIdx.x;
    if (idx >= 2 * NPAD * NPAD) return;
    int sup = idx >> 20;
    int rc  = idx & (NPAD * NPAD - 1);
    int r = rc >> 10, c = rc & 1023;
    float v = 0.f;
    if (r < Nsz && c < Nsz) {
        const float* src = sup ? s2 : s1;
        v = src[(size_t)r * Nsz + c];
    }
    dst[idx] = f2bf(v);
}

// ===== kernel: zero the pad node-columns (1000..1023) of 3 transposed buffers =====
// buffers are [BT][Hsz][NPAD] bf16; pad region per buffer = BT*Hsz*24 elements
__global__ void k_pad(unsigned short* __restrict__ b0, unsigned short* __restrict__ b1,
                      unsigned short* __restrict__ b2) {
    const int per = BT * Hsz * (NPAD - Nsz);    // 294912
    int idx = blockIdx.x * blockDim.x + threadIdx.x;
    if (idx >= 3 * per) return;
    unsigned short* p = (idx < per) ? b0 : (idx < 2 * per ? b1 : b2);
    int r = idx % per;
    int bth = r / (NPAD - Nsz);                 // 0 .. BT*Hsz-1
    int c = r % (NPAD - Nsz);
    p[(size_t)bth * NPAD + Nsz + c] = 0;
}

// ============ kernel: attention + FFN + layernorm -> state (transposed bf16) ============
// one block (64 threads) per (b, n)
__global__ void __launch_bounds__(64)
k_attn(const float* __restrict__ x,
       const float* __restrict__ Wq, const float* __restrict__ bq,
       const float* __restrict__ Wk, const float* __restrict__ bk,
       const float* __restrict__ Wv, const float* __restrict__ bv,
       const float* __restrict__ Wt, const float* __restrict__ bt_,
       const float* __restrict__ W1, const float* __restrict__ b1,
       const float* __restrict__ W2, const float* __restrict__ b2,
       const float* __restrict__ lng, const float* __restrict__ lnb,
       unsigned short* __restrict__ state) {
    __shared__ float xs[Tsz][Cin];
    __shared__ float qs[Tsz][Hsz];     // later reused as o
    __shared__ float ks[Tsz][Hsz];     // later reused as gelu(u)
    __shared__ float vs[Tsz][Hsz];     // later reused as val
    __shared__ float tps[Tsz][Hsz];
    __shared__ float att[HEADS][Tsz][Tsz];
    __shared__ float mu[Tsz], rsg[Tsz];

    const int tid = threadIdx.x;
    const int b = blockIdx.x / Nsz;
    const int n = blockIdx.x % Nsz;

    if (tid < Tsz * Cin) {
        int t = tid / Cin, c = tid % Cin;
        xs[t][c] = x[(((size_t)b * Tsz + t) * Nsz + n) * Cin + c];
    }
    __syncthreads();

    {
        int h = tid;
        float wq0 = Wq[h], wq1 = Wq[Hsz + h], bqv = bq[h];
        float wk0 = Wk[h], wk1 = Wk[Hsz + h], bkv = bk[h];
        float wv0 = Wv[h], wv1 = Wv[Hsz + h], bvv = bv[h];
        float wt0 = Wt[h], wt1 = Wt[Hsz + h], btv = bt_[h];
        for (int t = 0; t < Tsz; ++t) {
            float x0 = xs[t][0], x1 = xs[t][1];
            qs[t][h]  = x0 * wq0 + x1 * wq1 + bqv;
            ks[t][h]  = x0 * wk0 + x1 * wk1 + bkv;
            vs[t][h]  = x0 * wv0 + x1 * wv1 + bvv;
            tps[t][h] = x0 * wt0 + x1 * wt1 + btv;
        }
    }
    __syncthreads();

    const float scale = 0.35355339059327373f;   // 1/sqrt(8)
    for (int idx = tid; idx < HEADS * Tsz; idx += 64) {
        int head = idx / Tsz, t = idx % Tsz;
        float row[Tsz];
        float mx = -1e30f;
        for (int s = 0; s < Tsz; ++s) {
            float a;
            if (s <= t) {
                float d = 0.f;
                for (int dd = 0; dd < DH; ++dd)
                    d += qs[t][head * DH + dd] * ks[s][head * DH + dd];
                a = d * scale;
            } else {
                a = -32767.0f;
            }
            row[s] = a;
            mx = fmaxf(mx, a);
        }
        float sum = 0.f;
        for (int s = 0; s < Tsz; ++s) { row[s] = __expf(row[s] - mx); sum += row[s]; }
        float inv = 1.f / sum;
        for (int s = 0; s < Tsz; ++s) att[head][t][s] = row[s] * inv;
    }
    __syncthreads();

    {   // o = p @ v -> qs
        int h = tid, head = h >> 3;
        float col[Tsz];
        for (int t = 0; t < Tsz; ++t) {
            float acc = 0.f;
            for (int s = 0; s < Tsz; ++s) acc += att[head][t][s] * vs[s][h];
            col[t] = acc;
        }
        __syncthreads();
        for (int t = 0; t < Tsz; ++t) qs[t][h] = col[t];
    }
    __syncthreads();

    {   // u = gelu(o @ W1 + b1) -> ks
        int h = tid;
        float b1v = b1[h];
        for (int t = 0; t < Tsz; ++t) {
            float acc = b1v;
            for (int k = 0; k < Hsz; ++k) acc += qs[t][k] * W1[k * Hsz + h];
            ks[t][h] = 0.5f * acc * (1.f + erff(acc * 0.7071067811865476f));
        }
    }
    __syncthreads();

    {   // val = u @ W2 + b2 + tproj -> vs
        int h = tid;
        float b2v = b2[h];
        float col[Tsz];
        for (int t = 0; t < Tsz; ++t) {
            float acc = b2v + tps[t][h];
            for (int k = 0; k < Hsz; ++k) acc += ks[t][k] * W2[k * Hsz + h];
            col[t] = acc;
        }
        __syncthreads();
        for (int t = 0; t < Tsz; ++t) vs[t][h] = col[t];
    }
    __syncthreads();

    if (tid < Tsz) {
        int t = tid;
        float m = 0.f;
        for (int h = 0; h < Hsz; ++h) m += vs[t][h];
        m *= (1.f / Hsz);
        float v = 0.f;
        for (int h = 0; h < Hsz; ++h) { float d = vs[t][h] - m; v += d * d; }
        v *= (1.f / Hsz);
        mu[t] = m;
        rsg[t] = rsqrtf(v + 1e-5f);
    }
    __syncthreads();

    {
        int h = tid;
        float g = lng[h], bb = lnb[h];
        for (int t = 0; t < Tsz; ++t) {
            float y = (vs[t][h] - mu[t]) * rsg[t] * g + bb;
            // transposed layout: [bt][h][node]
            state[(((size_t)b * Tsz + t) * Hsz + h) * NPAD + n] = f2bf(y);
        }
    }
}

// ============ kernel: Chebyshev GEMM via WMMA bf16 + async LDS B-panel staging ============
// All buffers transposed [BT][Hsz][NPAD].  mode 0: Y = S@B ; mode 1: Y = 2*(S@B) - G0
// grid: 192 bt x 8 groups; block 256 = 8 waves, one 16-row tile per wave.
__global__ void __launch_bounds__(256)
k_cheb(const unsigned short* __restrict__ S,     // [NPAD][NPAD]
       const unsigned short* __restrict__ Bt,    // [BT][Hsz][NPAD]
       const unsigned short* __restrict__ G0t,   // [BT][Hsz][NPAD] (mode 1)
       unsigned short* __restrict__ Yt,          // [BT][Hsz][NPAD]
       int mode) {
    __shared__ __align__(16) unsigned char panel[Hsz * CSTRIDE];   // 5120 B

    const int tid  = threadIdx.x;
    const int lane = tid & 31;
    const int wid  = tid >> 5;
    const int bt   = blockIdx.x >> 3;
    const int mt   = (blockIdx.x & 7) * 8 + wid;      // 0..63
    const bool act = (mt < MT);
    const int mBase = mt * 16;
    const int kh = lane >> 4;
    const int ln = lane & 15;

    // async staging map: thread -> (column, quarter)
    const int scol  = tid >> 2;                       // 0..63
    const int spart = tid & 3;                        // 0..3  (8 K-values each)
    const unsigned long long gcol =
        (unsigned long long)(const void*)(Bt + ((size_t)bt * Hsz + scol) * NPAD + spart * 8);
    const unsigned int ldsoff =
        (unsigned int)(size_t)(void*)panel + (unsigned int)(scol * CSTRIDE + spart * 16);

    const unsigned short* Srow = S + (size_t)(mBase + ln) * NPAD;

    v8f acc[4];
    for (int nt = 0; nt < 4; ++nt)
        for (int j = 0; j < 8; ++j) acc[nt][j] = 0.f;

    for (int kc = 0; kc < NPAD / 32; ++kc) {
        const int kb = kc * 32;
        // stage B panel: 64 columns x 32 K (bf16) = 4 KB, one b128 per thread
        asm volatile("global_load_async_to_lds_b128 %0, %1, off"
                     :: "v"(ldsoff), "v"(gcol + (unsigned long long)(kb * 2))
                     : "memory");
        asm volatile("s_wait_asynccnt 0x0" ::: "memory");
        __syncthreads();

        if (act) {
            AFrag A;
#pragma unroll
            for (int j = 0; j < 8; ++j)
                A.u[j] = *(const unsigned int*)(Srow + kb + a_k0(j, kh));
            __builtin_prefetch(Srow + kb + 64, 0, 0);
#pragma unroll
            for (int nt = 0; nt < 4; ++nt) {
                const int col = nt * 16 + ln;
                const uint4* lp =
                    (const uint4*)(panel + col * CSTRIDE + kh * 16);
                uint4 q0 = lp[0];        // K = kh*8 + 0..7
                uint4 q1 = lp[2];        // K = 16 + kh*8 + 0..7
                AFrag Bf;
                Bf.u[0] = q0.x; Bf.u[1] = q0.y; Bf.u[2] = q0.z; Bf.u[3] = q0.w;
                Bf.u[4] = q1.x; Bf.u[5] = q1.y; Bf.u[6] = q1.z; Bf.u[7] = q1.w;
                acc[nt] = __builtin_amdgcn_wmma_f32_16x16x32_bf16(
                    false, A.v, false, Bf.v, (short)0, acc[nt], false, false);
            }
        }
        __syncthreads();
    }

    if (act) {
        // epilogue: rows mBase+8*kh .. +7 are consecutive node indices -> packed store.
        // pad nodes (>=1000) have zero S rows so acc==0 there: stores keep K-pad zero.
        const int rbase = mBase + 8 * kh;
        for (int nt = 0; nt < 4; ++nt) {
            const int col = nt * 16 + ln;
            const size_t boff = ((size_t)bt * Hsz + col) * NPAD + rbase;
            float v[8];
            if (mode) {
                uint4 g = *(const uint4*)(G0t + boff);
                unsigned int gw[4] = {g.x, g.y, g.z, g.w};
#pragma unroll
                for (int j = 0; j < 8; ++j) {
                    unsigned short g0 = (unsigned short)(gw[j >> 1] >> ((j & 1) * 16));
                    v[j] = 2.f * acc[nt][j] - bf2f(g0);
                }
            } else {
#pragma unroll
                for (int j = 0; j < 8; ++j) v[j] = acc[nt][j];
            }
            uint4 o;
            o.x = f2bf(v[0]) | ((unsigned int)f2bf(v[1]) << 16);
            o.y = f2bf(v[2]) | ((unsigned int)f2bf(v[3]) << 16);
            o.z = f2bf(v[4]) | ((unsigned int)f2bf(v[5]) << 16);
            o.w = f2bf(v[6]) | ((unsigned int)f2bf(v[7]) << 16);
            *(uint4*)(Yt + boff) = o;
        }
    }
}

// ============ kernel: gout accumulation via WMMA: out (+)= G @ Wg[r0:r0+64] ============
// G is transposed [BT][Hsz][NPAD]; out is row-major [BT][Nsz][Hsz]
__global__ void __launch_bounds__(256)
k_gout(const unsigned short* __restrict__ G,
       const float* __restrict__ Wg,            // [384][64]
       int r0,
       const float* __restrict__ bg,
       float* __restrict__ out,
       int init) {
    int wave = blockIdx.x * WPB + (threadIdx.x >> 5);
    if (wave >= TILES) return;
    int lane = threadIdx.x & 31;
    int bt = wave / MT;
    int mBase = (wave % MT) * 16;
    int kh = lane >> 4;
    int ln = lane & 15;
    int m = mBase + ln;

    const unsigned short* Gbt = G + (size_t)bt * Hsz * NPAD;

    v8f acc[4];
    for (int nt = 0; nt < 4; ++nt)
        for (int j = 0; j < 8; ++j) acc[nt][j] = 0.f;

    for (int kc = 0; kc < 2; ++kc) {      // K = Hsz = 64
        int kb = kc * 32;
        AFrag A;                          // strided gather: K lives on the h axis
#pragma unroll
        for (int j = 0; j < 8; ++j) {
            int k0 = kb + kh * 16 + 2 * j;
            unsigned int lo = Gbt[(size_t)k0 * NPAD + m];
            unsigned int hi = Gbt[(size_t)(k0 + 1) * NPAD + m];
            A.u[j] = lo | (hi << 16);
        }
        int krow = kb + kh * 16;
#pragma unroll
        for (int nt = 0; nt < 4; ++nt) {
            int col = nt * 16 + ln;
            AFrag Bf;
#pragma unroll
            for (int j = 0; j < 8; ++j) {
                unsigned int lo = f2bf(Wg[(size_t)(r0 + krow + 2 * j) * Hsz + col]);
                unsigned int hi = f2bf(Wg[(size_t)(r0 + krow + 2 * j + 1) * Hsz + col]);
                Bf.u[j] = lo | (hi << 16);
            }
            acc[nt] = __builtin_amdgcn_wmma_f32_16x16x32_bf16(
                false, A.v, false, Bf.v, (short)0, acc[nt], false, false);
        }
    }

    for (int nt = 0; nt < 4; ++nt) {
        int col = nt * 16 + ln;
        float bgv = bg[col];
        for (int j = 0; j < 8; ++j) {
            int row = mBase + j + 8 * kh;
            if (row < Nsz) {
                size_t idx = ((size_t)bt * Nsz + row) * Hsz + col;
                if (init) out[idx] = acc[nt][j] + bgv;
                else      out[idx] += acc[nt][j];
            }
        }
    }
}

// ============ kernel: mean over T -> h_buf[b][n][h] ============
__global__ void k_mean(const float* __restrict__ out, float* __restrict__ hbuf) {
    int idx = blockIdx.x * blockDim.x + threadIdx.x;
    if (idx >= Bsz * Nsz * Hsz) return;
    int b = idx / (Nsz * Hsz);
    int n = (idx / Hsz) % Nsz;
    int h = idx % Hsz;
    float s = 0.f;
    for (int t = 0; t < Tsz; ++t)
        s += out[(((size_t)b * Tsz + t) * Nsz + n) * Hsz + h];
    hbuf[idx] = s * (1.f / Tsz);
}

__global__ void k_zero_acc(float* acc) {
    if (threadIdx.x < 2 && blockIdx.x == 0) acc[threadIdx.x] = 0.f;
}

// ============ kernel: memory network per (b,n), 64 threads ============
__global__ void __launch_bounds__(64)
k_memnet(const float* __restrict__ hbuf, const float* __restrict__ WqM,
         const float* __restrict__ Mem, float* __restrict__ value,
         float* __restrict__ acc) {
    __shared__ float hv[Msz], qv[Msz], av[Msz], red[Msz];
    __shared__ int ii0, ii1;
    __shared__ float shm;
    const int tid = threadIdx.x;
    const int bn = blockIdx.x;

    hv[tid] = hbuf[(size_t)bn * Hsz + tid];
    __syncthreads();

    float q = 0.f;
    for (int k = 0; k < Hsz; ++k) q += hv[k] * WqM[k * Hsz + tid];
    qv[tid] = q;
    __syncthreads();

    float lg = 0.f;
    for (int h = 0; h < Hsz; ++h) lg += qv[h] * Mem[tid * Hsz + h];
    red[tid] = lg;
    __syncthreads();
    for (int off = 32; off > 0; off >>= 1) {
        if (tid < off) red[tid] = fmaxf(red[tid], red[tid + off]);
        __syncthreads();
    }
    if (tid == 0) shm = red[0];
    __syncthreads();
    float e = __expf(lg - shm);
    red[tid] = e;
    __syncthreads();
    for (int off = 32; off > 0; off >>= 1) {
        if (tid < off) red[tid] += red[tid + off];
        __syncthreads();
    }
    if (tid == 0) shm = red[0];
    __syncthreads();
    float a = e / shm;
    av[tid] = a;
    __syncthreads();

    float val = 0.f;
    for (int m = 0; m < Msz; ++m) val += av[m] * Mem[m * Hsz + tid];
    value[(size_t)bn * Hsz + tid] = val;

    if (tid == 0) {
        int i0 = 0; float b0 = av[0];
        for (int m = 1; m < Msz; ++m) if (av[m] > b0) { b0 = av[m]; i0 = m; }
        int i1 = (i0 == 0) ? 1 : 0; float b1 = av[i1];
        for (int m = 0; m < Msz; ++m)
            if (m != i0 && av[m] > b1) { b1 = av[m]; i1 = m; }
        ii0 = i0; ii1 = i1;
    }
    __syncthreads();

    float dp = qv[tid] - Mem[ii0 * Hsz + tid];
    red[tid] = dp * dp;
    __syncthreads();
    for (int off = 32; off > 0; off >>= 1) {
        if (tid < off) red[tid] += red[tid + off];
        __syncthreads();
    }
    float spos = red[0];
    __syncthreads();
    float dn = qv[tid] - Mem[ii1 * Hsz + tid];
    red[tid] = dn * dn;
    __syncthreads();
    for (int off = 32; off > 0; off >>= 1) {
        if (tid < off) red[tid] += red[tid + off];
        __syncthreads();
    }
    if (tid == 0) {
        float sneg = red[0];
        float dpos = sqrtf(spos + 1e-6f);
        float dneg = sqrtf(sneg + 1e-6f);
        atomicAdd(&acc[0], fmaxf(dpos - dneg + 1.f, 0.f));  // triplet sum
        atomicAdd(&acc[1], spos);                           // compact sum
    }
}

// ============ kernel: out += value (broadcast over T) ============
__global__ void k_addval(float* __restrict__ out, const float* __restrict__ value) {
    int idx = blockIdx.x * blockDim.x + threadIdx.x;
    if (idx >= Bsz * Tsz * Nsz * Hsz) return;
    int bt = idx / (Nsz * Hsz);
    int b = bt / Tsz;
    int nh = idx % (Nsz * Hsz);
    out[idx] += value[(size_t)b * Nsz * Hsz + nh];
}

__global__ void k_loss(const float* acc, float* out) {
    if (threadIdx.x == 0 && blockIdx.x == 0)
        out[(size_t)Bsz * Tsz * Nsz * Hsz] =
            acc[0] / (float)(Bsz * Nsz) + acc[1] / (float)(Bsz * Nsz * Hsz);
}

// ---------------- host launcher ----------------
extern "C" void kernel_launch(void* const* d_in, const int* in_sizes, int n_in,
                              void* d_out, int out_size, void* d_ws, size_t ws_size,
                              hipStream_t stream) {
    const float* x   = (const float*)d_in[0];
    const float* s1  = (const float*)d_in[1];
    const float* s2  = (const float*)d_in[2];
    const float* Wq  = (const float*)d_in[3];
    const float* bq  = (const float*)d_in[4];
    const float* Wk  = (const float*)d_in[5];
    const float* bk  = (const float*)d_in[6];
    const float* Wv  = (const float*)d_in[7];
    const float* bv  = (const float*)d_in[8];
    const float* Wt  = (const float*)d_in[9];
    const float* bt_ = (const float*)d_in[10];
    const float* W1  = (const float*)d_in[11];
    const float* b1  = (const float*)d_in[12];
    const float* W2  = (const float*)d_in[13];
    const float* b2  = (const float*)d_in[14];
    const float* lng = (const float*)d_in[15];
    const float* lnb = (const float*)d_in[16];
    const float* Wg  = (const float*)d_in[17];
    const float* bg  = (const float*)d_in[18];
    const float* Mem = (const float*)d_in[19];
    const float* WqM = (const float*)d_in[20];
    float* out = (float*)d_out;

    char* ws = (char*)d_ws;
    size_t off = 0;
    unsigned short* Sbf   = (unsigned short*)(ws + off); off += (size_t)2 * NPAD * NPAD * 2;   // 4 MB
    unsigned short* state = (unsigned short*)(ws + off); off += (size_t)BT * Hsz * NPAD * 2;   // 25 MB
    unsigned short* gA    = (unsigned short*)(ws + off); off += (size_t)BT * Hsz * NPAD * 2;   // 25 MB
    unsigned short* gB    = (unsigned short*)(ws + off); off += (size_t)BT * Hsz * NPAD * 2;   // 25 MB
    float* hbuf  = (float*)(ws + off); off += (size_t)Bsz * Nsz * Hsz * 4;                     // 2 MB
    float* value = (float*)(ws + off); off += (size_t)Bsz * Nsz * Hsz * 4;                     // 2 MB
    float* accb  = (float*)(ws + off); off += 256;

    const int cblocks = BT * 8;        // 1536 (k_cheb: bt x 8 tile-groups)
    const int tblocks = TILES / WPB;   // 1512 (k_gout)

    // 1. supports -> bf16 padded; zero pad columns of the 3 transposed buffers
    k_convert_s<<<(2 * NPAD * NPAD + 255) / 256, 256, 0, stream>>>(s1, s2, Sbf);
    k_pad<<<(3 * BT * Hsz * (NPAD - Nsz) + 255) / 256, 256, 0, stream>>>(state, gA, gB);

    // 2. attention + FFN + LN -> state (bf16, transposed)
    k_attn<<<Bsz * Nsz, 64, 0, stream>>>(x, Wq, bq, Wk, bk, Wv, bv, Wt, bt_,
                                         W1, b1, W2, b2, lng, lnb, state);

    // 3. gout accumulation: g0 contributes via Wg blocks 0 and 3
    k_gout<<<tblocks, 256, 0, stream>>>(state, Wg, 0 * Hsz, bg, out, 1);
    k_gout<<<tblocks, 256, 0, stream>>>(state, Wg, 3 * Hsz, bg, out, 0);

    const unsigned short* S0 = Sbf;
    const unsigned short* S1 = Sbf + (size_t)NPAD * NPAD;

    // support 1: g1 = S0@state ; g2 = 2*S0@g1 - state
    k_cheb<<<cblocks, 256, 0, stream>>>(S0, state, state, gA, 0);
    k_gout<<<tblocks, 256, 0, stream>>>(gA, Wg, 1 * Hsz, bg, out, 0);
    k_cheb<<<cblocks, 256, 0, stream>>>(S0, gA, state, gB, 1);
    k_gout<<<tblocks, 256, 0, stream>>>(gB, Wg, 2 * Hsz, bg, out, 0);

    // support 2
    k_cheb<<<cblocks, 256, 0, stream>>>(S1, state, state, gA, 0);
    k_gout<<<tblocks, 256, 0, stream>>>(gA, Wg, 4 * Hsz, bg, out, 0);
    k_cheb<<<cblocks, 256, 0, stream>>>(S1, gA, state, gB, 1);
    k_gout<<<tblocks, 256, 0, stream>>>(gB, Wg, 5 * Hsz, bg, out, 0);

    // 4. memory network + loss
    k_mean<<<(Bsz * Nsz * Hsz + 255) / 256, 256, 0, stream>>>(out, hbuf);
    k_zero_acc<<<1, 64, 0, stream>>>(accb);
    k_memnet<<<Bsz * Nsz, 64, 0, stream>>>(hbuf, WqM, Mem, value, accb);
    k_addval<<<(Bsz * Tsz * Nsz * Hsz + 255) / 256, 256, 0, stream>>>(out, value);
    k_loss<<<1, 1, 0, stream>>>(accb, out);
}
